// Attention_27986006901419
// MI455X (gfx1250) — matrix operand order repaired and checked by reading, not compiled, not run on hardware
//
#include <hip/hip_runtime.h>
#include <cstdint>
#include <cstddef>

// ---------------------------------------------------------------------------
// Problem constants (from reference): B=2, S=2048, D=4096, H=32, KVH=8,
// HD=128, WIN=1024. All GEMMs run as bf16 WMMA (v_wmma_f32_16x16x32_bf16),
// f32 accumulate. Flash-attention with online softmax over the causal window.
// Tile staging uses the CDNA5 Tensor Data Mover (tensor_load_to_lds) with
// hardware LDS row padding, double-buffered and pipelined via TENSORcnt.
// ---------------------------------------------------------------------------
#define BATCH   2
#define SEQ     2048
#define DMODEL  4096
#define NH      32
#define NKVH    8
#define HDIM    128
#define WINSZ   1024
#define MROWS   (BATCH * SEQ)     // 4096
#define QDIM    (NH * HDIM)       // 4096
#define KVDIM   (NKVH * HDIM)     // 1024

typedef unsigned short u16;
typedef __attribute__((ext_vector_type(16))) __bf16 v16bf;
typedef __attribute__((ext_vector_type(8)))  float  v8f;
typedef __attribute__((ext_vector_type(4)))  unsigned int u32x4;
typedef __attribute__((ext_vector_type(8)))  unsigned int u32x8;

__device__ __forceinline__ u16 f32_to_bf16(float f) {
  unsigned u = __float_as_uint(f);
  unsigned r = 0x7FFFu + ((u >> 16) & 1u);   // round-to-nearest-even
  return (u16)((u + r) >> 16);
}
__device__ __forceinline__ float bf16_to_f32(u16 h) {
  return __uint_as_float(((unsigned)h) << 16);
}

union FragU { uint4 q[2]; v16bf v; };

// Two 16B chunks -> one 16x32 / 32x16 bf16 fragment (8 VGPRs).
__device__ __forceinline__ v16bf load_frag2(const u16* p0, const u16* p1) {
  FragU u;
  u.q[0] = *(const uint4*)p0;
  u.q[1] = *(const uint4*)p1;
  return u.v;
}

__device__ __forceinline__ float rmax16(float v) {
#pragma unroll
  for (int i = 1; i < 16; i <<= 1) v = fmaxf(v, __shfl_xor(v, i, 32));
  return v;
}
__device__ __forceinline__ float rsum16(float v) {
#pragma unroll
  for (int i = 1; i < 16; i <<= 1) v += __shfl_xor(v, i, 32);
  return v;
}

// ---------------------------------------------------------------------------
// Tensor Data Mover: 2-D bf16 tile, global -> LDS, with optional LDS padding.
// D# group 0 (4 SGPRs): count=1 | lds_addr | global_addr(57b) | type=2.
// D# group 1 (8 SGPRs): data_size=2B, pad cfg, tensor dims, tile dims, stride.
// One issue per wave; completion tracked with TENSORcnt.
// padcfg: 0 (no pad) or (1<<20)|(interval_code<<22)|(amount_code<<25).
// ---------------------------------------------------------------------------
#define PADCFG_16DW_4DW ((1u << 20) | (3u << 22) | (3u << 25))  // rows of 16 DW + 4 DW pad
#define PADCFG_64DW_4DW ((1u << 20) | (5u << 22) | (3u << 25))  // rows of 64 DW + 4 DW pad

__device__ __forceinline__ void tdm_load_2d(
    unsigned lds_byte_addr, const u16* gptr,
    unsigned tile_w, unsigned tile_h, unsigned row_stride_elems,
    unsigned padcfg) {
  unsigned long long ga = (unsigned long long)(uintptr_t)gptr;
  unsigned long long s0 = (unsigned long long)row_stride_elems;   // dim0 stride (48b)

  u32x4 g0;
  g0[0] = 1u;                                             // count=1, user mode
  g0[1] = lds_byte_addr;                                  // lds_addr
  g0[2] = (unsigned)(ga & 0xFFFFFFFFu);                   // global_addr[31:0]
  g0[3] = (unsigned)((ga >> 32) & 0x01FFFFFFu) | (2u << 30);  // addr[56:32] | type=2

  u32x8 g1;
  g1[0] = (1u << 16) | padcfg;                            // data_size=1 (2B) | pad cfg
  g1[1] = (tile_w & 0xFFFFu) << 16;                       // tensor_dim0 @ bit 48
  g1[2] = (tile_w >> 16) | ((tile_h & 0xFFFFu) << 16);    // dim0 hi | tensor_dim1 lo
  g1[3] = (tile_h >> 16) | ((tile_w & 0xFFFFu) << 16);    // dim1 hi | tile_dim0
  g1[4] = (tile_h & 0xFFFFu);                             // tile_dim1 (tile_dim2 = 0)
  g1[5] = (unsigned)(s0 & 0xFFFFFFFFu);                   // tensor_dim0_stride lo
  g1[6] = (unsigned)((s0 >> 32) & 0xFFFFu);               // stride hi (dim1 stride = 0)
  g1[7] = 0u;

  asm volatile("tensor_load_to_lds %0, %1" :: "s"(g0), "s"(g1) : "memory");
}

__device__ __forceinline__ unsigned lds_addr_of(const void* p) {
  return (unsigned)(uintptr_t)p;   // low 32 bits of a generic LDS pointer = LDS offset
}

// ---------------------------------------------------------------------------
// f32 -> bf16 conversion (grid-stride)
// ---------------------------------------------------------------------------
__global__ void cvt_f32_bf16(const float* __restrict__ src, u16* __restrict__ dst, int n) {
  int i = blockIdx.x * blockDim.x + threadIdx.x;
  int stride = gridDim.x * blockDim.x;
  for (; i < n; i += stride) dst[i] = f32_to_bf16(src[i]);
}

// ---------------------------------------------------------------------------
// GEMM:  Y[M,N] = X[M,K] * W[N,K]^T   (bf16 inputs, f32 accumulate)
// Block = 256 threads (8 waves), tile 128x128, K-step 32.
// Wave tile 64x32 -> 8 WMMA accumulators (4 M-tiles x 2 N-tiles).
// A/B tiles staged by the TDM into double-buffered padded LDS; wave 0 issues
// the DMA for tile t+1 before computing tile t (s_wait_tensorcnt 2 pipelining).
// Writes bf16 (Yb) or f32 (Yf), whichever pointer is non-null.
// ---------------------------------------------------------------------------
#define BM 128
#define BN 128
#define BK 32
#define LDT 40   // padded LDS row stride: 32 bf16 (16 DW) + 4 DW TDM pad -> 80B rows

__global__ __launch_bounds__(256) void gemm_bf16_nt(
    const u16* __restrict__ X, const u16* __restrict__ W,
    u16* __restrict__ Yb, float* __restrict__ Yf, int M, int N, int K) {
  __shared__ u16 As[2][BM][LDT];
  __shared__ u16 Bs[2][BM][LDT];

  const int tid  = threadIdx.x;
  const int lane = tid & 31;
  const int wid  = tid >> 5;
  const int m0 = blockIdx.y * BM;
  const int n0 = blockIdx.x * BN;
  const int wrow = (wid & 1) * 64;        // wave row offset   (2 wave rows)
  const int wcol = (wid >> 1) * 32;       // wave col offset   (4 wave cols)
  const int half = lane >> 4;             // 0 / 1
  const int l16  = lane & 15;
  const bool issuer = (wid == 0);

  v8f acc[4][2];
#pragma unroll
  for (int mt = 0; mt < 4; ++mt)
#pragma unroll
    for (int nt = 0; nt < 2; ++nt) acc[mt][nt] = {};

  const u16* xrow = X + (size_t)m0 * K;
  const u16* wrowp = W + (size_t)n0 * K;

  if (issuer) {
    tdm_load_2d(lds_addr_of(&As[0][0][0]), xrow,  BK, BM, K, PADCFG_16DW_4DW);
    tdm_load_2d(lds_addr_of(&Bs[0][0][0]), wrowp, BK, BM, K, PADCFG_16DW_4DW);
  }

  const int nk = K / BK;
  for (int t = 0; t < nk; ++t) {
    const int cur = t & 1;
    if (issuer) {
      if (t + 1 < nk) {
        const int k0 = (t + 1) * BK;
        tdm_load_2d(lds_addr_of(&As[cur ^ 1][0][0]), xrow + k0,  BK, BM, K, PADCFG_16DW_4DW);
        tdm_load_2d(lds_addr_of(&Bs[cur ^ 1][0][0]), wrowp + k0, BK, BM, K, PADCFG_16DW_4DW);
        __builtin_amdgcn_s_wait_tensorcnt((short)2);   // tile t landed
      } else {
        __builtin_amdgcn_s_wait_tensorcnt((short)0);   // last tile landed
      }
    }
    __syncthreads();

    // A fragment: lane row = l16, K chunks {half*8 .. +7} and {half*8+16 .. +23}
    v16bf afrag[4];
#pragma unroll
    for (int mt = 0; mt < 4; ++mt) {
      const u16* ar = &As[cur][wrow + mt * 16 + l16][0];
      afrag[mt] = load_frag2(ar + half * 8, ar + half * 8 + 16);
    }
#pragma unroll
    for (int nt = 0; nt < 2; ++nt) {
      // B fragment: lane col = l16, K chunks {half*8 .. +7} and {half*8+16 .. +23}
      const u16* br = &Bs[cur][wcol + nt * 16 + l16][half * 8];
      v16bf bfrag = load_frag2(br, br + 16);
#pragma unroll
      for (int mt = 0; mt < 4; ++mt)
        acc[mt][nt] = __builtin_amdgcn_wmma_f32_16x16x32_bf16(
            false, afrag[mt], false, bfrag, (short)0, acc[mt][nt], false, false);
    }
    __syncthreads();   // all waves done reading buf[cur] before it is re-filled
  }

  // C/D layout: element e -> row = e + 8*half, col = l16
#pragma unroll
  for (int mt = 0; mt < 4; ++mt)
#pragma unroll
    for (int nt = 0; nt < 2; ++nt)
#pragma unroll
      for (int e = 0; e < 8; ++e) {
        int gr = m0 + wrow + mt * 16 + e + 8 * half;
        int gc = n0 + wcol + nt * 16 + l16;
        float val = acc[mt][nt][e];
        if (Yf) Yf[(size_t)gr * N + gc] = val;
        else    Yb[(size_t)gr * N + gc] = f32_to_bf16(val);
      }
}

// ---------------------------------------------------------------------------
// RoPE (interleaved pairs within each head) + optional scale (for Q: 1/sqrt(HD))
// t: bf16 [rows][ncols], rows ordered b*S+s
// ---------------------------------------------------------------------------
__global__ void rope_kernel(u16* __restrict__ t, const float* __restrict__ cosp,
                            const float* __restrict__ sinp, int rows, int ncols,
                            float scale) {
  const int npairs_row = ncols / 2;
  const long total = (long)rows * npairs_row;
  long i = (long)blockIdx.x * blockDim.x + threadIdx.x;
  long stride = (long)gridDim.x * blockDim.x;
  for (; i < total; i += stride) {
    int row = (int)(i / npairs_row);
    int cp  = (int)(i % npairs_row);
    int hd2 = cp & (HDIM / 2 - 1);                 // pair index within head
    int col0 = (cp / (HDIM / 2)) * HDIM + 2 * hd2; // head base + 2*i
    int s = row & (SEQ - 1);                       // sequence position
    float c = cosp[s * (HDIM / 2) + hd2];
    float sn = sinp[s * (HDIM / 2) + hd2];
    u16* p = t + (size_t)row * ncols + col0;
    float x0 = bf16_to_f32(p[0]);
    float x1 = bf16_to_f32(p[1]);
    p[0] = f32_to_bf16((x0 * c - x1 * sn) * scale);
    p[1] = f32_to_bf16((x0 * sn + x1 * c) * scale);
  }
}

// ---------------------------------------------------------------------------
// Flash attention. grid = (S/64, H, B), block = 128 (4 waves).
// Each wave owns 16 query rows; the block shares the same key window, so
// block-wide barriers are uniform. Per 32-key block:
//   K block staged by the TDM (padded rows), V staged transposed by hand,
//   scores (2 tiles x 4 WMMA over HD) -> mask -> online softmax ->
//   P via LDS re-layout -> 8 WMMA accumulating the 16x128 output tile.
// ---------------------------------------------------------------------------
#define LKS 136   // 128 bf16 (64 DW) + 4 DW TDM pad
#define LVS 40    // 32 + 8 bf16 pad

__global__ __launch_bounds__(128) void attn_kernel(
    const u16* __restrict__ q, const u16* __restrict__ k,
    const u16* __restrict__ v, u16* __restrict__ ctx) {
  __shared__ u16 Ks[32][LKS];          // [key][hd]   row-major (TDM-staged)
  __shared__ u16 Vt[HDIM][LVS];        // [hd][key]   transposed
  __shared__ u16 Ps[4][16][LVS];       // per-wave P tile [qrow][key]

  const int qt = blockIdx.x, h = blockIdx.y, b = blockIdx.z;
  const int q0 = qt * 64;
  const int tid  = threadIdx.x;
  const int lane = tid & 31;
  const int wid  = tid >> 5;
  const int half = lane >> 4;
  const int l16  = lane & 15;
  const int kvh  = h >> 2;             // H/KVH = 4

  const u16* qbase = q + (size_t)b * SEQ * QDIM + (size_t)h * HDIM;
  const u16* kbase = k + (size_t)b * SEQ * KVDIM + (size_t)kvh * HDIM;
  const u16* vbase = v + (size_t)b * SEQ * KVDIM + (size_t)kvh * HDIM;

  // Q fragments for this wave's 16 rows, 4 HD chunks of 32 (direct global).
  const int qr0 = q0 + wid * 16;
  v16bf qfrag[4];
#pragma unroll
  for (int c = 0; c < 4; ++c) {
    const u16* qr = qbase + (size_t)(qr0 + l16) * QDIM + c * 32 + half * 8;
    qfrag[c] = load_frag2(qr, qr + 16);
  }

  v8f oacc[8];
#pragma unroll
  for (int nt = 0; nt < 8; ++nt) oacc[nt] = {};
  float m[8], rs[8];
#pragma unroll
  for (int r = 0; r < 8; ++r) { m[r] = -3.0e38f; rs[r] = 0.0f; }

  int lo = q0 + 63 - (WINSZ - 1);
  if (lo < 0) lo = 0;
  lo &= ~31;                            // window length is a multiple of 32
  const int hi = q0 + 63;

  for (int kb = lo; kb <= hi; kb += 32) {
    // ---- Stage K via TDM (wave 0), V transposed by all threads ----
    if (wid == 0)
      tdm_load_2d(lds_addr_of(&Ks[0][0]), kbase + (size_t)kb * KVDIM,
                  HDIM, 32, KVDIM, PADCFG_64DW_4DW);
    {
      int key = tid & 31;
      int hc  = (tid >> 5) * 32;
      const u16* vrow = vbase + (size_t)(kb + key) * KVDIM;
#pragma unroll
      for (int i = 0; i < 4; ++i) {
        union { uint4 qv; u16 s[8]; } u;
        u.qv = *(const uint4*)&vrow[hc + i * 8];
#pragma unroll
        for (int j = 0; j < 8; ++j) Vt[hc + i * 8 + j][key] = u.s[j];
      }
    }
    if (wid == 0) __builtin_amdgcn_s_wait_tensorcnt((short)0);
    __syncthreads();

    // ---- Scores for two 16-key tiles + online softmax ----
#pragma unroll
    for (int t16 = 0; t16 < 2; ++t16) {
      v8f sacc = {};
#pragma unroll
      for (int c = 0; c < 4; ++c) {
        // B frag: col = key (l16), K = hd chunks {half*8..+7} and {half*8+16..+23}
        const u16* br = &Ks[t16 * 16 + l16][c * 32 + half * 8];
        v16bf bfrag = load_frag2(br, br + 16);
        sacc = __builtin_amdgcn_wmma_f32_16x16x32_bf16(
            false, qfrag[c], false, bfrag, (short)0, sacc, false, false);
      }
#pragma unroll
      for (int r = 0; r < 8; ++r) {
        int row = qr0 + r + 8 * half;
        int col = kb + t16 * 16 + l16;
        bool ok = (col <= row) && (row - col < WINSZ);
        float sv = ok ? sacc[r] : -1.0e9f;
        float tmax = rmax16(sv);
        float mn = fmaxf(m[r], tmax);
        float al = __expf(m[r] - mn);
        float p  = (sv < -5.0e8f) ? 0.0f : __expf(sv - mn);
        rs[r] = rs[r] * al + rsum16(p);
        m[r]  = mn;
#pragma unroll
        for (int nt = 0; nt < 8; ++nt) oacc[nt][r] *= al;
        Ps[wid][r + 8 * half][t16 * 16 + l16] = f32_to_bf16(p);
      }
    }
    __syncthreads();   // P tile fully written before fragment re-load

    // ---- P (A-frag) x V (B-frag from transposed LDS) ----
    {
      const u16* pr = &Ps[wid][l16][half * 8];
      v16bf pfrag = load_frag2(pr, pr + 16);
#pragma unroll
      for (int nt = 0; nt < 8; ++nt) {
        const u16* vr = &Vt[nt * 16 + l16][half * 8];
        v16bf vfrag = load_frag2(vr, vr + 16);
        oacc[nt] = __builtin_amdgcn_wmma_f32_16x16x32_bf16(
            false, pfrag, false, vfrag, (short)0, oacc[nt], false, false);
      }
    }
    __syncthreads();   // done reading Ks/Vt/Ps before next stage
  }

  // ---- Normalize and write ctx (bf16) ----
#pragma unroll
  for (int r = 0; r < 8; ++r) {
    int row = qr0 + r + 8 * half;
    float inv = (rs[r] > 0.0f) ? (1.0f / rs[r]) : 0.0f;
#pragma unroll
    for (int nt = 0; nt < 8; ++nt) {
      int col = h * HDIM + nt * 16 + l16;
      ctx[((size_t)b * SEQ + row) * QDIM + col] = f32_to_bf16(oacc[nt][r] * inv);
    }
  }
}

// ---------------------------------------------------------------------------
// Host-side orchestration
// ---------------------------------------------------------------------------
extern "C" void kernel_launch(void* const* d_in, const int* in_sizes, int n_in,
                              void* d_out, int out_size, void* d_ws, size_t ws_size,
                              hipStream_t stream) {
  const float* x    = (const float*)d_in[0];
  const float* wq   = (const float*)d_in[1];
  const float* wk   = (const float*)d_in[2];
  const float* wv   = (const float*)d_in[3];
  const float* wo   = (const float*)d_in[4];
  const float* cosp = (const float*)d_in[5];
  const float* sinp = (const float*)d_in[6];
  // d_in[7] = mask (computed analytically on-device), d_in[8] = positions (implicit)

  // Workspace layout (u16 elements) — ~201 MB total
  u16* ws = (u16*)d_ws;
  const size_t XB  = 0;
  const size_t WQB = XB  + (size_t)MROWS * DMODEL;
  const size_t WKB = WQB + (size_t)QDIM  * DMODEL;
  const size_t WVB = WKB + (size_t)KVDIM * DMODEL;
  const size_t WOB = WVB + (size_t)KVDIM * DMODEL;
  const size_t QB  = WOB + (size_t)DMODEL * QDIM;
  const size_t KB  = QB  + (size_t)MROWS * QDIM;
  const size_t VB  = KB  + (size_t)MROWS * KVDIM;
  const size_t CTX = VB  + (size_t)MROWS * KVDIM;
  (void)ws_size; (void)in_sizes; (void)n_in; (void)out_size;

  const int T = 256;
  auto blocks = [](long n, int t) { long b = (n + t - 1) / t; return (unsigned)(b > 16384 ? 16384 : b); };

  // 1) f32 -> bf16 conversions
  cvt_f32_bf16<<<blocks((long)MROWS * DMODEL, T), T, 0, stream>>>(x,  ws + XB,  MROWS * DMODEL);
  cvt_f32_bf16<<<blocks((long)QDIM  * DMODEL, T), T, 0, stream>>>(wq, ws + WQB, QDIM  * DMODEL);
  cvt_f32_bf16<<<blocks((long)KVDIM * DMODEL, T), T, 0, stream>>>(wk, ws + WKB, KVDIM * DMODEL);
  cvt_f32_bf16<<<blocks((long)KVDIM * DMODEL, T), T, 0, stream>>>(wv, ws + WVB, KVDIM * DMODEL);
  cvt_f32_bf16<<<blocks((long)DMODEL * QDIM,  T), T, 0, stream>>>(wo, ws + WOB, DMODEL * QDIM);

  // 2) Q/K/V projections (bf16 WMMA GEMMs, bf16 outputs)
  gemm_bf16_nt<<<dim3(QDIM / BN,  MROWS / BM), 256, 0, stream>>>(
      ws + XB, ws + WQB, ws + QB, nullptr, MROWS, QDIM, DMODEL);
  gemm_bf16_nt<<<dim3(KVDIM / BN, MROWS / BM), 256, 0, stream>>>(
      ws + XB, ws + WKB, ws + KB, nullptr, MROWS, KVDIM, DMODEL);
  gemm_bf16_nt<<<dim3(KVDIM / BN, MROWS / BM), 256, 0, stream>>>(
      ws + XB, ws + WVB, ws + VB, nullptr, MROWS, KVDIM, DMODEL);

  // 3) RoPE (scale 1/sqrt(HD) folded into Q)
  const float scale = 0.088388347648318447f;  // 128^-0.5
  rope_kernel<<<blocks((long)MROWS * QDIM / 2,  T), T, 0, stream>>>(
      ws + QB, cosp, sinp, MROWS, QDIM, scale);
  rope_kernel<<<blocks((long)MROWS * KVDIM / 2, T), T, 0, stream>>>(
      ws + KB, cosp, sinp, MROWS, KVDIM, 1.0f);

  // 4) Windowed-causal flash attention -> ctx (bf16)
  attn_kernel<<<dim3(SEQ / 64, NH, BATCH), 128, 0, stream>>>(
      ws + QB, ws + KB, ws + VB, ws + CTX);

  // 5) Output projection -> f32 result
  gemm_bf16_nt<<<dim3(DMODEL / BN, MROWS / BM), 256, 0, stream>>>(
      ws + CTX, ws + WOB, nullptr, (float*)d_out, MROWS, DMODEL, DMODEL);
}
